// KT_58360015618320
// MI455X (gfx1250) — matrix operand order repaired and checked by reading, not compiled, not run on hardware
//
// MI455X / gfx1250 implementation of the AKT-style transformer reference.
//
// Roofline: ~80 GFLOP of GEMMs vs ~300 MB HBM traffic (score tensors fused
// on-chip, never materialized to DRAM). Compute-dominated -> all matmuls go
// through v_wmma_f32_16x16x32_bf16 (f32 inputs downconverted to bf16 operands,
// f32 accumulation). Attention is fused per (b,h,16-query-row) block in one
// wave32 using LDS for the 16x1024 score rows; softmax/cumsum/decay done with
// wave shuffles; both QK^T and PV einsums use WMMA.

#include <hip/hip_runtime.h>
#include <hip/hip_bf16.h>
#include <math.h>

typedef __attribute__((ext_vector_type(16))) __bf16 bf16x16;
typedef __attribute__((ext_vector_type(8)))  float  f32x8;

#define BS   8
#define SEQ  1024
#define DMODEL 256
#define NHEAD  8
#define DK   32
#define ROWS (BS * SEQ)   // 8192

// ---------------------------------------------------------------------------
// WMMA fragment helpers (wave32; A and B both use contiguous k = 16*(lane/16) + element)
// ---------------------------------------------------------------------------

__device__ inline f32x8 wmma_bf16(bf16x16 a, bf16x16 b, f32x8 c) {
  return __builtin_amdgcn_wmma_f32_16x16x32_bf16(false, a, false, b,
                                                 (short)0, c, false, false);
}

// A (16x32 bf16, MxK), stored row-major with leading dim `ld` (elements).
// lanes 0-15: M=lane, K=0..15; lanes 16-31: M=lane-16, K=16..31
__device__ inline bf16x16 load_a_frag(const __bf16* p, int ld) {
  const int lane = threadIdx.x & 31;
  const int m = lane & 15;
  const int kb = (lane < 16) ? 0 : 16;
  const __bf16* row = p + m * ld;
  bf16x16 a;
#pragma unroll
  for (int v = 0; v < 4; ++v) {
    a[2 * v]     = row[kb + 2 * v];
    a[2 * v + 1] = row[kb + 2 * v + 1];
  }
#pragma unroll
  for (int v = 0; v < 4; ++v) {
    a[8 + 2 * v]     = row[8 + kb + 2 * v];
    a[8 + 2 * v + 1] = row[8 + kb + 2 * v + 1];
  }
  return a;
}

// B (32x16 bf16, KxN), stored row-major [K][ld].
// lanes 0-15: N=lane, K=0..15 in v0..7 (2/VGPR); lanes 16-31: N=lane-16, K=16..31
__device__ inline bf16x16 load_b_frag(const __bf16* p, int ld) {
  const int lane = threadIdx.x & 31;
  const int n = lane & 15;
  const int kb = (lane < 16) ? 0 : 16;
  bf16x16 b;
#pragma unroll
  for (int v = 0; v < 8; ++v) {
    b[2 * v]     = p[(kb + 2 * v) * ld + n];
    b[2 * v + 1] = p[(kb + 2 * v + 1) * ld + n];
  }
  return b;
}

// ---------------------------------------------------------------------------
// Wave32 reduction / scan helpers
// ---------------------------------------------------------------------------

__device__ inline float wave_max(float v) {
#pragma unroll
  for (int o = 16; o > 0; o >>= 1) v = fmaxf(v, __shfl_xor(v, o));
  return v;
}
__device__ inline float wave_sum(float v) {
#pragma unroll
  for (int o = 16; o > 0; o >>= 1) v += __shfl_xor(v, o);
  return v;
}
__device__ inline float wave_scan_inc(float v) {
  const int lane = threadIdx.x & 31;
#pragma unroll
  for (int o = 1; o < 32; o <<= 1) {
    float u = __shfl_up(v, o);
    if (lane >= o) v += u;
  }
  return v;
}

// ---------------------------------------------------------------------------
// Generic WMMA GEMM: C[M,N] = act(A[M,K] @ W[K,N] + bias)
// block = 256 threads (8 waves); tile 128(M) x 64(N), K-step 32.
// M % 128 == 0, N % 64 == 0, K % 32 == 0 (all shapes here satisfy this).
// ---------------------------------------------------------------------------

template <int ACT>  // 0 = none, 1 = relu
__global__ __launch_bounds__(256) void gemm_kernel(
    const float* __restrict__ A, const float* __restrict__ W,
    const float* __restrict__ bias, float* __restrict__ C,
    int M, int N, int K) {
  __shared__ __bf16 As[128][40];  // 32 cols + pad
  __shared__ __bf16 Bs[32][72];   // 64 cols + pad

  const int tid = threadIdx.x;
  const int lane = tid & 31;
  const int wid = tid >> 5;
  const int m0 = blockIdx.y * 128;
  const int n0 = blockIdx.x * 64;

  f32x8 acc[4];
#pragma unroll
  for (int i = 0; i < 4; ++i) acc[i] = (f32x8){0, 0, 0, 0, 0, 0, 0, 0};

  for (int k0 = 0; k0 < K; k0 += 32) {
    {  // A tile: each thread a contiguous half-row of 16
      const int r = tid >> 1;
      const int c = (tid & 1) * 16;
      const float* src = A + (size_t)(m0 + r) * K + k0 + c;
#pragma unroll
      for (int e = 0; e < 16; ++e) As[r][c + e] = (__bf16)src[e];
    }
    {  // B tile: each thread 8 contiguous
      const int r = tid >> 3;
      const int c = (tid & 7) * 8;
      const float* src = W + (size_t)(k0 + r) * N + n0 + c;
#pragma unroll
      for (int e = 0; e < 8; ++e) Bs[r][c + e] = (__bf16)src[e];
    }
    if (k0 + 32 < K) {  // prefetch next K tiles (global_prefetch_b8)
      __builtin_prefetch(A + (size_t)(m0 + (tid >> 1)) * K + k0 + 32, 0, 1);
      __builtin_prefetch(W + (size_t)(k0 + 32 + (tid >> 3)) * N + n0, 0, 1);
    }
    __syncthreads();

    bf16x16 af = load_a_frag(&As[wid * 16][0], 40);
#pragma unroll
    for (int nt = 0; nt < 4; ++nt) {
      bf16x16 bfg = load_b_frag(&Bs[0][nt * 16], 72);
      acc[nt] = wmma_bf16(af, bfg, acc[nt]);
    }
    __syncthreads();
  }

#pragma unroll
  for (int nt = 0; nt < 4; ++nt) {
    const int col = n0 + nt * 16 + (lane & 15);
    const float bv = bias ? bias[col] : 0.f;
#pragma unroll
    for (int r = 0; r < 8; ++r) {
      const int row = m0 + wid * 16 + ((lane < 16) ? r : r + 8);
      float v = acc[nt][r] + bv;
      if (ACT == 1) v = fmaxf(v, 0.f);
      C[(size_t)row * N + col] = v;
    }
  }
}

// ---------------------------------------------------------------------------
// Fused attention. One wave per (16 query rows, head, batch).
// PK: q/k projection [ROWS, 256] (head h at cols h*32..), PV: v projection.
// Implements: softmax1(masked) -> cumsum -> distance decay -> softmax2
// (unmasked, faithful to reference) -> optional zero_pad -> P @ V.
// Dynamic LDS: S[16][1024] f32 | P[16][1024] bf16 | tmp[1024] f32 | Q[16][32] bf16
// ---------------------------------------------------------------------------

__global__ __launch_bounds__(32) void attn_kernel(
    const float* __restrict__ PK, const float* __restrict__ PV,
    const float* __restrict__ gam, float* __restrict__ O, int mask_k) {
  extern __shared__ char smem_raw[];
  float*  S   = (float*)smem_raw;                                   // 64 KB
  __bf16* P   = (__bf16*)(smem_raw + 16 * 1024 * 4);                // 32 KB
  float*  tmp = (float*)(smem_raw + 16 * 1024 * 4 + 16 * 1024 * 2); //  4 KB
  __bf16* Q   = (__bf16*)(tmp + 1024);                              //  1 KB

  const int lane = threadIdx.x;
  const int i0 = blockIdx.x * 16;
  const int h  = blockIdx.y;
  const int b  = blockIdx.z;
  const size_t baseRow = (size_t)b * SEQ;
  const float inv_scale = 0.17677669529663687f;  // 1/sqrt(DK)

  // Q tile 16x32 -> LDS bf16
  {
    const int r = lane >> 1;
    const int c0 = (lane & 1) * 16;
    const float* src = PK + (baseRow + i0 + r) * DMODEL + h * DK + c0;
#pragma unroll
    for (int e = 0; e < 16; ++e) Q[r * 32 + c0 + e] = (__bf16)src[e];
  }
  __syncthreads();
  const bf16x16 qf = load_a_frag(Q, 32);

  // scores[16,1024] = (Q @ K^T) / sqrt(dk); one WMMA per 16-col tile (K=32)
  for (int jt = 0; jt < 64; ++jt) {
    const int n = lane & 15;
    const int kb = (lane < 16) ? 0 : 16;
    const float* kr = PK + (baseRow + jt * 16 + n) * DMODEL + h * DK;
    bf16x16 bfg;
#pragma unroll
    for (int v = 0; v < 8; ++v) {
      bfg[2 * v]     = (__bf16)kr[kb + 2 * v];
      bfg[2 * v + 1] = (__bf16)kr[kb + 2 * v + 1];
    }
    f32x8 c = (f32x8){0, 0, 0, 0, 0, 0, 0, 0};
    c = wmma_bf16(qf, bfg, c);
#pragma unroll
    for (int r = 0; r < 8; ++r) {
      const int row = (lane < 16) ? r : r + 8;
      S[row * 1024 + jt * 16 + n] = c[r] * inv_scale;
    }
  }
  __syncthreads();

  const float g = gam[h];
  const float gamma = -log1pf(__expf(g));  // -softplus

  for (int r = 0; r < 16; ++r) {
    const int i = i0 + r;
    float* Sr = S + r * 1024;
    int maskend = i + mask_k;
    if (maskend > SEQ) maskend = SEQ;

    // pass A: masked max
    float m = -3.0e38f;
#pragma unroll 4
    for (int e = 0; e < 32; ++e) {
      const int j = e * 32 + lane;
      const float sc = Sr[j];
      if (j < maskend) m = fmaxf(m, sc);
    }
    m = wave_max(m);

    // pass B: unnormalized softmax1 weights + inclusive cumsum (ordered)
    float carry = 0.f;
    for (int e = 0; e < 32; ++e) {
      const int j = e * 32 + lane;
      const float w = (j < maskend) ? __expf(Sr[j] - m) : 0.f;
      const float sc = wave_scan_inc(w);
      tmp[j] = carry + sc;
      carry += __shfl(sc, 31);  // chunk total
    }
    const float sumW = carry;
    const float invSumW = (sumW > 0.f) ? 1.f / sumW : 0.f;

    // pass C: dist = sqrt(clip((1 - cum/sum) * |i-j|)); eff = clamp(exp(g*d));
    // modulated scores overwrite S; track max for softmax2 (UNMASKED).
    float m2 = -3.0e38f;
#pragma unroll 4
    for (int e = 0; e < 32; ++e) {
      const int j = e * 32 + lane;
      float rem = (invSumW == 0.f) ? 0.f : fmaxf(1.f - tmp[j] * invSumW, 0.f);
      const float pos = fabsf((float)(i - j));
      const float dist = sqrtf(rem * pos);
      float eff = __expf(gamma * dist);
      eff = fminf(fmaxf(eff, 1e-5f), 1e5f);
      const float val = Sr[j] * eff;
      Sr[j] = val;
      m2 = fmaxf(m2, val);
    }
    m2 = wave_max(m2);

    // pass D: exp + sum
    float sumE = 0.f;
#pragma unroll 4
    for (int e = 0; e < 32; ++e) {
      const int j = e * 32 + lane;
      const float p = __expf(Sr[j] - m2);
      Sr[j] = p;
      sumE += p;
    }
    sumE = wave_sum(sumE);
    const float invE = 1.f / sumE;
    const bool zero = (mask_k == 0) && (i == 0);  // zero_pad row
#pragma unroll 4
    for (int e = 0; e < 32; ++e) {
      const int j = e * 32 + lane;
      P[r * 1024 + j] = (__bf16)(zero ? 0.f : Sr[j] * invE);
    }
  }
  __syncthreads();

  // O tile = P[16,1024] @ V[1024,32]; 32 K-steps x 2 N-halves
  f32x8 acc0 = (f32x8){0, 0, 0, 0, 0, 0, 0, 0};
  f32x8 acc1 = (f32x8){0, 0, 0, 0, 0, 0, 0, 0};
  const int n = lane & 15;
  const int kb = (lane < 16) ? 0 : 16;
  for (int kt = 0; kt < 32; ++kt) {
    const bf16x16 af = load_a_frag(P + kt * 32, 1024);
    bf16x16 b0, b1;
#pragma unroll
    for (int v = 0; v < 8; ++v) {
      const float* r0 = PV + (baseRow + kt * 32 + kb + 2 * v) * DMODEL + h * DK;
      const float* r1 = PV + (baseRow + kt * 32 + kb + 2 * v + 1) * DMODEL + h * DK;
      b0[2 * v]     = (__bf16)r0[n];
      b0[2 * v + 1] = (__bf16)r1[n];
      b1[2 * v]     = (__bf16)r0[n + 16];
      b1[2 * v + 1] = (__bf16)r1[n + 16];
    }
    acc0 = wmma_bf16(af, b0, acc0);
    acc1 = wmma_bf16(af, b1, acc1);
  }
#pragma unroll
  for (int r = 0; r < 8; ++r) {
    const int row = (lane < 16) ? r : r + 8;
    float* dst = O + (baseRow + i0 + row) * DMODEL + h * DK;
    dst[n] = acc0[r];
    dst[n + 16] = acc1[r];
  }
}

// ---------------------------------------------------------------------------
// Residual + LayerNorm: out = LN(x + y). One wave per 256-wide row.
// ---------------------------------------------------------------------------

__global__ __launch_bounds__(256) void add_ln_kernel(
    const float* __restrict__ X, const float* __restrict__ Y,
    const float* __restrict__ g, const float* __restrict__ bet,
    float* __restrict__ Out) {
  const int lane = threadIdx.x & 31;
  const int wid = threadIdx.x >> 5;
  const size_t row = (size_t)blockIdx.x * 8 + wid;
  const float* x = X + row * DMODEL;
  const float* y = Y + row * DMODEL;
  float v[8];
  float s = 0.f;
#pragma unroll
  for (int e = 0; e < 8; ++e) {
    v[e] = x[e * 32 + lane] + y[e * 32 + lane];
    s += v[e];
  }
  s = wave_sum(s);
  const float mean = s * (1.f / 256.f);
  float q = 0.f;
#pragma unroll
  for (int e = 0; e < 8; ++e) {
    const float d = v[e] - mean;
    q += d * d;
  }
  q = wave_sum(q);
  const float inv = rsqrtf(q * (1.f / 256.f) + 1e-5f);
  float* o = Out + row * DMODEL;
#pragma unroll
  for (int e = 0; e < 8; ++e) {
    const int c = e * 32 + lane;
    o[c] = g[c] * (v[e] - mean) * inv + bet[c];
  }
}

// out = base + diff_param[pid[row]] * diff   (elementwise over [8192,256])
__global__ void combine_embed_kernel(const float* __restrict__ base,
                                     const float* __restrict__ diff,
                                     const float* __restrict__ diff_param,
                                     const int* __restrict__ pid,
                                     float* __restrict__ out) {
  const int idx = blockIdx.x * 256 + threadIdx.x;
  const int row = idx >> 8;
  const float p = diff_param[pid[row]];
  out[idx] = base[idx] + p * diff[idx];
}

// concat([x, qe], -1) -> [8192, 512]
__global__ void concat_kernel(const float* __restrict__ x,
                              const float* __restrict__ qe,
                              float* __restrict__ out) {
  const int idx = blockIdx.x * 256 + threadIdx.x;
  const int row = idx >> 9;
  const int c = idx & 511;
  out[idx] = (c < 256) ? x[(size_t)row * 256 + c]
                       : qe[(size_t)row * 256 + (c - 256)];
}

// sigmoid(h[8192,256] @ w[256] + b)
__global__ void head_kernel(const float* __restrict__ hsrc,
                            const float* __restrict__ w,
                            const float* __restrict__ b,
                            float* __restrict__ out) {
  const int row = blockIdx.x * 256 + threadIdx.x;
  const float* hr = hsrc + (size_t)row * 256;
  float acc = 0.f;
#pragma unroll 8
  for (int k = 0; k < 256; ++k) acc += hr[k] * w[k];
  acc += b[0];
  out[row] = 1.f / (1.f + __expf(-acc));
}

// ---------------------------------------------------------------------------
// Host orchestration.
// Input leaf order (setup_inputs dict insertion order, pytrees recursed):
//  0 q_data  1 qa_data
//  2 Wqe 3 bqe  4 Wqae 5 bqae  6 Wqd 7 bqd  8 Wqad 9 bqad  10 diff_param
//  11+15*l: Wk bk Wv bv Wo bo gam ln1g ln1b W1 b1 W2 b2 ln2g ln2b (l=0..2)
//  56 Wf1 57 bf1 58 Wf2 59 bf2 60 Wf3 61 bf3
//  62 pid_data
// ---------------------------------------------------------------------------

extern "C" void kernel_launch(void* const* d_in, const int* in_sizes, int n_in,
                              void* d_out, int out_size, void* d_ws,
                              size_t ws_size, hipStream_t stream) {
  (void)in_sizes; (void)n_in; (void)out_size; (void)ws_size;

  const float* q_data  = (const float*)d_in[0];
  const float* qa_data = (const float*)d_in[1];
  const float* Wqe  = (const float*)d_in[2];
  const float* bqe  = (const float*)d_in[3];
  const float* Wqae = (const float*)d_in[4];
  const float* bqae = (const float*)d_in[5];
  const float* Wqd  = (const float*)d_in[6];
  const float* bqd  = (const float*)d_in[7];
  const float* Wqad = (const float*)d_in[8];
  const float* bqad = (const float*)d_in[9];
  const float* diffp = (const float*)d_in[10];
  auto LP = [&](int l, int o) { return (const float*)d_in[11 + 15 * l + o]; };
  const float* Wf1 = (const float*)d_in[56];
  const float* bf1 = (const float*)d_in[57];
  const float* Wf2 = (const float*)d_in[58];
  const float* bf2 = (const float*)d_in[59];
  const float* Wf3 = (const float*)d_in[60];
  const float* bf3 = (const float*)d_in[61];
  const int* pid = (const int*)d_in[62];

  // workspace carve (floats): A,B,C,D,F = 8192*256 each; E = 8192*2048
  float* ws = (float*)d_ws;
  const size_t SZ = (size_t)ROWS * DMODEL;  // 2,097,152
  float* A  = ws;            // q_emb  (persists to concat)
  float* Bq = ws + SZ;       // qa_emb / layer0 x1
  float* Cy = ws + 2 * SZ;   // y
  float* Dx = ws + 3 * SZ;   // x
  float* F  = ws + 4 * SZ;   // ffn output
  float* E  = ws + 5 * SZ;   // 16.8M floats: proj scratch / ffn hidden / head
  float* E0 = E;             // projK (q and k share Wk: kq_same)
  float* E1 = E + SZ;        // projV
  float* E2 = E + 2 * SZ;    // attention out
  float* E3 = E + 3 * SZ;    // mha out

  const int ATTN_SHMEM = 16 * 1024 * 4 + 16 * 1024 * 2 + 1024 * 4 + 16 * 32 * 2;

  auto gemm = [&](const float* a, const float* w, const float* bias, float* c,
                  int M, int N, int K) {
    gemm_kernel<0><<<dim3(N / 64, M / 128), 256, 0, stream>>>(a, w, bias, c, M, N, K);
  };
  auto gemm_relu = [&](const float* a, const float* w, const float* bias,
                       float* c, int M, int N, int K) {
    gemm_kernel<1><<<dim3(N / 64, M / 128), 256, 0, stream>>>(a, w, bias, c, M, N, K);
  };

  // --- embeddings ---
  gemm(q_data, Wqe, bqe, E0, ROWS, DMODEL, 512);
  gemm(q_data, Wqd, bqd, E1, ROWS, DMODEL, 512);
  combine_embed_kernel<<<ROWS, 256, 0, stream>>>(E0, E1, diffp, pid, A);
  gemm(qa_data, Wqae, bqae, E0, ROWS, DMODEL, 1024);
  gemm(qa_data, Wqad, bqad, E1, ROWS, DMODEL, 1024);
  combine_embed_kernel<<<ROWS, 256, 0, stream>>>(E0, E1, diffp, pid, Bq);

  auto run_layer = [&](int l, const float* query, const float* value,
                       float* x1, float* dst, int mask_k, bool ffn) {
    gemm(query, LP(l, 0), LP(l, 1), E0, ROWS, DMODEL, DMODEL);  // q==k proj (Wk)
    gemm(value, LP(l, 2), LP(l, 3), E1, ROWS, DMODEL, DMODEL);  // v proj (Wv)
    attn_kernel<<<dim3(SEQ / 16, NHEAD, BS), 32, ATTN_SHMEM, stream>>>(
        E0, E1, LP(l, 6), E2, mask_k);
    gemm(E2, LP(l, 4), LP(l, 5), E3, ROWS, DMODEL, DMODEL);     // Wo
    add_ln_kernel<<<ROWS / 8, 256, 0, stream>>>(query, E3, LP(l, 7), LP(l, 8), x1);
    if (ffn) {
      gemm_relu(x1, LP(l, 9), LP(l, 10), E, ROWS, 2048, DMODEL);  // ffn hidden
      gemm(E, LP(l, 11), LP(l, 12), F, ROWS, DMODEL, 2048);
      add_ln_kernel<<<ROWS / 8, 256, 0, stream>>>(x1, F, LP(l, 13), LP(l, 14), dst);
    }
  };

  run_layer(0, Bq, Bq, Bq, Cy, 1, true);   // blocks_1     -> y
  run_layer(1, A,  A,  Dx, Dx, 1, false);  // blocks_2[0]  -> x (A preserved)
  run_layer(2, Dx, Cy, Dx, Dx, 0, true);   // blocks_2[1]  -> x

  // --- output head ---
  float* Gc = E;               // concat [8192,512]
  float* Hc = E + 2 * SZ;      // fc1 out [8192,512]
  float* F2 = E + 4 * SZ;      // fc2 out [8192,256]
  concat_kernel<<<(ROWS * 512) / 256, 256, 0, stream>>>(Dx, A, Gc);
  gemm_relu(Gc, Wf1, bf1, Hc, ROWS, 512, 512);
  gemm_relu(Hc, Wf2, bf2, F2, ROWS, 256, 512);
  head_kernel<<<ROWS / 256, 256, 0, stream>>>(F2, Wf3, bf3, (float*)d_out);
}
